// ARDecoder_63307817943770
// MI455X (gfx1250) — compile-verified, matrix operand order checked by reading, not run on hardware
//
#include <hip/hip_runtime.h>
#include <cstdint>

// ---------------- types ----------------
typedef __attribute__((ext_vector_type(16))) __bf16 v16bf;
typedef __attribute__((ext_vector_type(8)))  float  v8f;
typedef __attribute__((ext_vector_type(4)))  int    v4i;
typedef __attribute__((ext_vector_type(4)))  float  v4f;

union frag_u { v4i i2[2]; v16bf v; };

// ---------------- problem dims ----------------
static constexpr int B_   = 64;
static constexpr int S_   = 512;
static constexpr int H_   = 1024;
static constexpr int E_   = 128;
static constexpr int L_   = 49;
static constexpr int EH   = E_ + H_;     // 1152
static constexpr int G3   = 3 * H_;      // 3072
static constexpr int ROWS = B_ * S_;     // 32768
static constexpr int KT_IH  = EH / 32;   // 36 k-tiles for gx GEMM
static constexpr int KT_HH  = H_ / 32;   // 32 k-tiles for recurrent GEMM
static constexpr int KT_OUT = H_ / 32;   // 32 k-tiles for logits GEMM
static constexpr int NWG_REC = 64;       // persistent WGs in recurrence (16 h-cols each)

// ---------------- workspace layout (bytes) ----------------
static constexpr size_t GX_OFF   = 0;
static constexpr size_t GX_BYTES = (size_t)ROWS * G3 * 4;                 // 402,653,184
static constexpr size_t WIH_OFF  = GX_OFF + GX_BYTES;
static constexpr size_t WIH_BYTES = (size_t)G3 * EH * 2;                  // 7,077,888
static constexpr size_t WHH_OFF  = WIH_OFF + WIH_BYTES;
static constexpr size_t WHH_BYTES = (size_t)G3 * H_ * 2;                  // 6,291,456
static constexpr size_t WOUT_OFF = WHH_OFF + WHH_BYTES;
static constexpr size_t WOUT_BYTES = (size_t)64 * H_ * 2;                 // 131,072 (L padded to 64)
static constexpr size_t OUTS_OFF = WOUT_OFF + WOUT_BYTES;
static constexpr size_t OUTS_BYTES = (size_t)ROWS * H_ * 2;               // 67,108,864
static constexpr size_t HF_OFF   = OUTS_OFF + OUTS_BYTES;                 // fp32 h ping-pong
static constexpr size_t HF_BYTES = (size_t)2 * B_ * H_ * 4;               // 524,288
static constexpr size_t HB_OFF   = HF_OFF + HF_BYTES;                     // bf16 h ping-pong
static constexpr size_t HB_BYTES = (size_t)2 * B_ * H_ * 2;               // 262,144
static constexpr size_t BAR_OFF  = HB_OFF + HB_BYTES;                     // 512 step-barrier slots
static constexpr size_t BAR_BYTES = 4096;
static constexpr size_t ZERO_BYTES = HF_BYTES + HB_BYTES + BAR_BYTES;     // contiguous zero region

// ---------------- CDNA5 async global->LDS path (guarded) ----------------
#if __has_builtin(__builtin_amdgcn_global_load_async_to_lds_b128)
#define HAVE_ASYNC_LDS 1
#endif

// copy 16 bytes (8 x bf16) global -> LDS; async (ASYNCcnt) when available.
// Builtin signature (from hipcc diagnostic): (v4i addrspace(1)*, v4i addrspace(3)*, imm, imm)
__device__ __forceinline__ void async_copy16(const unsigned short* g, unsigned short* l) {
#if defined(HAVE_ASYNC_LDS)
  __builtin_amdgcn_global_load_async_to_lds_b128(
      (__attribute__((address_space(1))) v4i*)(v4i*)(void*)g,
      (__attribute__((address_space(3))) v4i*)(v4i*)(void*)l, 0, 0);
#else
  *(v4i*)l = *(const v4i*)g;
#endif
}
__device__ __forceinline__ void wait_async0() {
#if defined(HAVE_ASYNC_LDS)
#if __has_builtin(__builtin_amdgcn_s_wait_asynccnt)
  __builtin_amdgcn_s_wait_asynccnt(0);
#else
  asm volatile("s_wait_asynccnt 0" ::: "memory");
#endif
#endif
}

// ---------------- device helpers ----------------
__device__ __forceinline__ unsigned short f2bf(float f) {
  unsigned int u = __builtin_bit_cast(unsigned int, f);
  u += 0x7fffu + ((u >> 16) & 1u);          // round-to-nearest-even
  return (unsigned short)(u >> 16);
}
// K index pattern of one lane element inside a 16-bit 16x32 A/B fragment
__device__ __forceinline__ int kpat(int half, int e) {
  return ((e < 8) ? e : (e + 8)) + half * 8;
}
__device__ __forceinline__ float sigmoidf_(float x) { return 1.0f / (1.0f + __expf(-x)); }
__device__ __forceinline__ float tanhf_(float x) {
  float cx = fminf(fmaxf(x, -15.0f), 15.0f);
  float e = __expf(2.0f * cx);
  return (e - 1.0f) / (e + 1.0f);
}
// IOBES transition constraint, computed analytically.
__device__ __forceinline__ bool iobes_allowed(int p, int q) {
  bool pOES = (p == 0) || (((p - 1) & 3) >= 2);   // O, E, S
  if (pOES) {
    if (q == 0) return true;
    int pos = (q - 1) & 3;
    return (pos == 0) || (pos == 3);              // B or S
  }
  if (q == 0) return false;                       // after B/I only I/E of same type
  int tp = (p - 1) >> 2, tq = (q - 1) >> 2, pos = (q - 1) & 3;
  return (tq == tp) && (pos == 1 || pos == 2);
}

// ---------------- weight pre-pack kernels (fragment-native layout) ----------------
__global__ __launch_bounds__(256) void pack_wih_kernel(const float* __restrict__ w,
                                                       unsigned short* __restrict__ dst) {
  int idx = blockIdx.x * 256 + threadIdx.x;
  if (idx >= (G3 / 16) * KT_IH * 32 * 16) return;
  int e = idx & 15, lane = (idx >> 4) & 31;
  int kt = (idx >> 9) % KT_IH;
  int nt = idx / (16 * 32 * KT_IH);
  int n = nt * 16 + (lane & 15);
  int k = kt * 32 + kpat(lane >> 4, e);
  dst[idx] = f2bf(w[n * EH + k]);
}

__global__ __launch_bounds__(256) void pack_whh_kernel(const float* __restrict__ w,
                                                       unsigned short* __restrict__ dst) {
  int idx = blockIdx.x * 256 + threadIdx.x;
  if (idx >= NWG_REC * 3 * KT_HH * 32 * 16) return;
  int e = idx & 15, lane = (idx >> 4) & 31;
  int kt = (idx >> 9) & 31;
  int g  = (idx >> 14) % 3;
  int wg = idx / (16 * 32 * KT_HH * 3);
  int n = g * H_ + wg * 16 + (lane & 15);
  int k = kt * 32 + kpat(lane >> 4, e);
  dst[idx] = f2bf(w[n * H_ + k]);
}

__global__ __launch_bounds__(256) void pack_wout_kernel(const float* __restrict__ w,
                                                        unsigned short* __restrict__ dst) {
  int idx = blockIdx.x * 256 + threadIdx.x;
  if (idx >= 4 * KT_OUT * 32 * 16) return;
  int e = idx & 15, lane = (idx >> 4) & 31;
  int kt = (idx >> 9) & 31;
  int nt = idx >> 14;
  int n = nt * 16 + (lane & 15);
  int k = kt * 32 + kpat(lane >> 4, e);
  dst[idx] = (n < L_) ? f2bf(w[n * H_ + k]) : (unsigned short)0;
}

__global__ __launch_bounds__(256) void zero_kernel(int* __restrict__ p, int n) {
  int i = blockIdx.x * 256 + threadIdx.x;
  if (i < n) p[i] = 0;
}

// ---------------- phase 1: gx = [emb[prev] ; word] @ w_ih^T + b_ih ----------------
// grid (48, 512). 128 thr = 4 waves. Double-buffered LDS A tile (built on VALU) and
// double-buffered LDS B tiles staged via async global->LDS, one barrier per k-tile.
__global__ __launch_bounds__(128) void gx_kernel(const float* __restrict__ word,
                                                 const int* __restrict__ labels,
                                                 const float* __restrict__ emb,
                                                 const unsigned short* __restrict__ wih_frag,
                                                 const float* __restrict__ b_ih,
                                                 float* __restrict__ gx) {
  // A: 64 rows x 32 cols bf16, row stride 40 (80B) -> conflict-free b128 reads
  __shared__ __attribute__((aligned(16))) unsigned short Ash[2][64 * 40];
  // B: 4 tiles, chunk-split layout [tile][chunk(2)][lane(32)][8] -> 16B lane stride
  __shared__ __attribute__((aligned(16))) unsigned short Bsh[2][4 * 2 * 32 * 8];
  const int tid = threadIdx.x;
  const int wave = tid >> 5, lane = tid & 31;
  const int col = lane & 15, half = lane >> 4;
  const int Mbase = blockIdx.y * 64, Nbase = blockIdx.x * 64;

  // per-thread A-build source (thread t builds row t/2, 16-col half (t&1))
  const int ar = tid >> 1, ac0 = (tid & 1) * 16;
  const int R = Mbase + ar;
  const int b = R >> 9, s = R & 511;
  const int prev = (s == 0) ? 0 : labels[b * S_ + s - 1];
  const float* emb_row  = emb + (size_t)prev * E_;
  const float* word_row = word + (size_t)R * H_;

  // wave w stages B tile nt==w: lane's 32B -> two 16B async chunks
  const unsigned short* bsrc0 =
      wih_frag + ((size_t)(blockIdx.x * 4 + wave) * KT_IH * 32 + lane) * 16;
  unsigned short* bdst0 = &Bsh[0][((wave * 2 + 0) * 32 + lane) * 8];
  unsigned short* bdst1 = &Bsh[0][((wave * 2 + 1) * 32 + lane) * 8];
  unsigned short* bdst0b = &Bsh[1][((wave * 2 + 0) * 32 + lane) * 8];
  unsigned short* bdst1b = &Bsh[1][((wave * 2 + 1) * 32 + lane) * 8];

  // prologue: stage B for kt=0
  async_copy16(bsrc0, bdst0);
  async_copy16(bsrc0 + 8, bdst1);

  v8f acc[4] = {};
  for (int kt = 0; kt < KT_IH; ++kt) {
    const int cur = kt & 1;
    // ---- build A tile (gather + f32->bf16) into Ash[cur] ----
    {
      const int k0 = kt * 32 + ac0;
      const float* src = (kt < 4) ? (emb_row + k0) : (word_row + (k0 - E_));
      alignas(16) unsigned short tmp[16];
      const v4f* s4 = (const v4f*)src;
#pragma unroll
      for (int q = 0; q < 4; ++q) {
        v4f f = s4[q];
        tmp[q * 4 + 0] = f2bf(f.x);
        tmp[q * 4 + 1] = f2bf(f.y);
        tmp[q * 4 + 2] = f2bf(f.z);
        tmp[q * 4 + 3] = f2bf(f.w);
      }
      unsigned short* dst = &Ash[cur][ar * 40 + ac0];
      *(v4i*)dst       = *(const v4i*)tmp;
      *(v4i*)(dst + 8) = *(const v4i*)(tmp + 8);
    }
    wait_async0();       // B[cur] landed in LDS
    __syncthreads();     // A[cur] + B[cur] visible; prev readers done with nxt buffers
    // ---- prefetch B for kt+1 into the other buffer (async, overlaps WMMAs) ----
    if (kt + 1 < KT_IH) {
      const unsigned short* src = bsrc0 + (size_t)(kt + 1) * 32 * 16;
      async_copy16(src, cur ? bdst0 : bdst0b);
      async_copy16(src + 8, cur ? bdst1 : bdst1b);
    }
    // ---- fragments: load everything, then issue all WMMAs ----
    const unsigned short* ap = &Ash[cur][(wave * 16 + col) * 40];
    frag_u A;
    A.i2[0] = *(const v4i*)(ap + half * 8);
    A.i2[1] = *(const v4i*)(ap + 16 + half * 8);
    frag_u Bf[4];
#pragma unroll
    for (int nt = 0; nt < 4; ++nt) {
      Bf[nt].i2[0] = *(const v4i*)&Bsh[cur][((nt * 2 + 0) * 32 + lane) * 8];
      Bf[nt].i2[1] = *(const v4i*)&Bsh[cur][((nt * 2 + 1) * 32 + lane) * 8];
    }
#pragma unroll
    for (int nt = 0; nt < 4; ++nt)
      acc[nt] = __builtin_amdgcn_wmma_f32_16x16x32_bf16(false, A.v, false, Bf[nt].v,
                                                        (short)0, acc[nt], false, false);
  }
  // epilogue: + b_ih, store fp32
#pragma unroll
  for (int nt = 0; nt < 4; ++nt) {
    const int cc = Nbase + nt * 16 + col;
    const float bias = b_ih[cc];
#pragma unroll
    for (int v = 0; v < 8; ++v) {
      const int row = Mbase + wave * 16 + v + 8 * half;
      gx[(size_t)row * G3 + cc] = acc[nt][v] + bias;
    }
  }
}

// ---------------- phase 2: persistent GRU recurrence ----------------
// 64 WGs x 128 thr. WG wg owns h columns [wg*16, wg*16+16). B (w_hh slice, 6MB bf16
// total) stays L2-resident; per-step gx slice is HBM-streamed -> prefetched up front.
__global__ __launch_bounds__(128) void recurrent_kernel(const float* __restrict__ gx,
                                                        const unsigned short* __restrict__ whh_frag,
                                                        const float* __restrict__ b_hh,
                                                        float* __restrict__ hf,
                                                        unsigned short* __restrict__ hb,
                                                        unsigned short* __restrict__ outs,
                                                        int* __restrict__ bar) {
  const int wg = blockIdx.x;
  const int tid = threadIdx.x;
  const int wave = tid >> 5, lane = tid & 31;
  const int col = lane & 15, half = lane >> 4;
  const int mrow0 = wave * 16;
  const int hc = wg * 16 + col;

  float bhh[3];
#pragma unroll
  for (int g = 0; g < 3; ++g) bhh[g] = b_hh[g * H_ + hc];

  for (int t = 0; t < S_; ++t) {
    // prefetch this step's gx epilogue operands (HBM stream) behind the GEMM
#pragma unroll
    for (int v = 0; v < 8; ++v) {
      const int b = mrow0 + v + 8 * half;
      const float* p = gx + ((size_t)b * S_ + t) * G3 + hc;
      __builtin_prefetch(p, 0, 1);
      __builtin_prefetch(p + H_, 0, 1);
      __builtin_prefetch(p + 2 * H_, 0, 1);
    }
    const unsigned short* hbr = hb + (size_t)(t & 1) * (B_ * H_);
    const float*          hfr = hf + (size_t)(t & 1) * (B_ * H_);
    v8f acc[3] = {};
    for (int kt = 0; kt < KT_HH; ++kt) {
      // A fragment: rows = batch, from bf16 h (L2-hot 128KB)
      const unsigned short* ap = hbr + (size_t)(mrow0 + col) * H_ + kt * 32;
      frag_u A;
      A.i2[0] = *(const v4i*)(ap + half * 8);
      A.i2[1] = *(const v4i*)(ap + 16 + half * 8);
      frag_u Bf[3];
#pragma unroll
      for (int g = 0; g < 3; ++g) {
        const unsigned short* bp =
            whh_frag + ((size_t)((wg * 3 + g) * KT_HH + kt) * 32 + lane) * 16;
        Bf[g].i2[0] = *(const v4i*)bp;
        Bf[g].i2[1] = *(const v4i*)(bp + 8);
      }
#pragma unroll
      for (int g = 0; g < 3; ++g)
        acc[g] = __builtin_amdgcn_wmma_f32_16x16x32_bf16(false, A.v, false, Bf[g].v,
                                                         (short)0, acc[g], false, false);
    }
    // gate nonlinearity + state update (r/z/n aligned per lane/vgpr)
    const int nb = (t + 1) & 1;
#pragma unroll
    for (int v = 0; v < 8; ++v) {
      const int b = mrow0 + v + 8 * half;     // batch row
      const float* gxp = gx + ((size_t)b * S_ + t) * G3;
      const float gr = gxp[hc], gz = gxp[H_ + hc], gn = gxp[2 * H_ + hc];
      const float ar = acc[0][v] + bhh[0];
      const float az = acc[1][v] + bhh[1];
      const float an = acc[2][v] + bhh[2];
      const float r = sigmoidf_(gr + ar);
      const float z = sigmoidf_(gz + az);
      const float n = tanhf_(gn + r * an);
      const float hp = hfr[(size_t)b * H_ + hc];
      const float hn = (1.0f - z) * n + z * hp;
      hf[(size_t)nb * (B_ * H_) + (size_t)b * H_ + hc] = hn;
      const unsigned short hbv = f2bf(hn);
      hb[(size_t)nb * (B_ * H_) + (size_t)b * H_ + hc] = hbv;
      outs[((size_t)b * S_ + t) * H_ + hc] = hbv;
    }
    // device-wide step barrier (per-step slot, no sense reversal needed)
    __threadfence();
    __syncthreads();
    if (tid == 0) {
      __hip_atomic_fetch_add(&bar[t], 1, __ATOMIC_RELEASE, __HIP_MEMORY_SCOPE_AGENT);
      while (__hip_atomic_load(&bar[t], __ATOMIC_ACQUIRE, __HIP_MEMORY_SCOPE_AGENT) < NWG_REC)
        __builtin_amdgcn_s_sleep(1);
    }
    __syncthreads();
  }
}

// ---------------- phase 3: logits = outs @ w_out^T + b_out, IOBES mask ----------------
__global__ __launch_bounds__(128) void logits_kernel(const unsigned short* __restrict__ outs,
                                                     const unsigned short* __restrict__ wout_frag,
                                                     const float* __restrict__ b_out,
                                                     const int* __restrict__ labels,
                                                     float* __restrict__ out) {
  const int tid = threadIdx.x;
  const int wave = tid >> 5, lane = tid & 31;
  const int col = lane & 15, half = lane >> 4;
  const int Mbase = blockIdx.x * 64;

  v8f acc[4] = {};
  const unsigned short* ap0 = outs + (size_t)(Mbase + wave * 16 + col) * H_;
  for (int kt = 0; kt < KT_OUT; ++kt) {
    const unsigned short* ap = ap0 + kt * 32;
    frag_u A;
    A.i2[0] = *(const v4i*)(ap + half * 8);
    A.i2[1] = *(const v4i*)(ap + 16 + half * 8);
    frag_u Bf[4];
#pragma unroll
    for (int nt = 0; nt < 4; ++nt) {
      const unsigned short* bp = wout_frag + ((size_t)(nt * KT_OUT + kt) * 32 + lane) * 16;
      Bf[nt].i2[0] = *(const v4i*)bp;
      Bf[nt].i2[1] = *(const v4i*)(bp + 8);
    }
#pragma unroll
    for (int nt = 0; nt < 4; ++nt)
      acc[nt] = __builtin_amdgcn_wmma_f32_16x16x32_bf16(false, A.v, false, Bf[nt].v,
                                                        (short)0, acc[nt], false, false);
  }
#pragma unroll
  for (int nt = 0; nt < 4; ++nt) {
    const int n = nt * 16 + col;
    if (n < L_) {
      const float bias = b_out[n];
#pragma unroll
      for (int v = 0; v < 8; ++v) {
        const int row = Mbase + wave * 16 + v + 8 * half;
        const int b = row >> 9, s = row & 511;
        const int prev = (s == 0) ? 0 : labels[b * S_ + s - 1];
        const float val = acc[nt][v] + bias;
        out[(size_t)row * L_ + n] = iobes_allowed(prev, n) ? val : -1e12f;
      }
    }
  }
}

// ---------------- host entry ----------------
extern "C" void kernel_launch(void* const* d_in, const int* in_sizes, int n_in,
                              void* d_out, int out_size, void* d_ws, size_t ws_size,
                              hipStream_t stream) {
  (void)in_sizes; (void)n_in; (void)out_size; (void)ws_size;
  const float* word   = (const float*)d_in[0];
  const int*   labels = (const int*)  d_in[1];
  const float* emb    = (const float*)d_in[2];
  const float* w_ih   = (const float*)d_in[3];
  const float* w_hh   = (const float*)d_in[4];
  const float* b_ih   = (const float*)d_in[5];
  const float* b_hh   = (const float*)d_in[6];
  const float* w_out  = (const float*)d_in[7];
  const float* b_out  = (const float*)d_in[8];

  char* ws = (char*)d_ws;
  float*          gx    = (float*)         (ws + GX_OFF);
  unsigned short* wihf  = (unsigned short*)(ws + WIH_OFF);
  unsigned short* whhf  = (unsigned short*)(ws + WHH_OFF);
  unsigned short* woutf = (unsigned short*)(ws + WOUT_OFF);
  unsigned short* outsb = (unsigned short*)(ws + OUTS_OFF);
  float*          hf    = (float*)         (ws + HF_OFF);
  unsigned short* hb    = (unsigned short*)(ws + HB_OFF);
  int*            bar   = (int*)           (ws + BAR_OFF);

  // weight pre-pack into WMMA-fragment layout + zero h/barrier state
  pack_wih_kernel <<<(G3 / 16) * KT_IH * 32 * 16 / 256, 256, 0, stream>>>(w_ih, wihf);
  pack_whh_kernel <<<NWG_REC * 3 * KT_HH * 32 * 16 / 256, 256, 0, stream>>>(w_hh, whhf);
  pack_wout_kernel<<<4 * KT_OUT * 32 * 16 / 256,          256, 0, stream>>>(w_out, woutf);
  {
    int zn = (int)(ZERO_BYTES / 4);
    zero_kernel<<<(zn + 255) / 256, 256, 0, stream>>>((int*)(ws + HF_OFF), zn);
  }
  // phase 1: big parallel GEMM for gate-input contributions
  gx_kernel<<<dim3(G3 / 64, ROWS / 64), 128, 0, stream>>>(word, labels, emb, wihf, b_ih, gx);
  // phase 2: persistent recurrence with per-step device barrier
  recurrent_kernel<<<NWG_REC, 128, 0, stream>>>(gx, whhf, b_hh, hf, hb, outsb, bar);
  // phase 3: output projection + IOBES transition mask
  logits_kernel<<<ROWS / 64, 128, 0, stream>>>(outsb, woutf, b_out, labels, (float*)d_out);
}
